// GraphAttnMultiHead_23021024707544
// MI455X (gfx1250) — matrix operand order repaired and checked by reading, not compiled, not run on hardware
//
#include <hip/hip_runtime.h>
#include <hip/hip_bf16.h>
#include <math.h>

// ---------------------------------------------------------------------------
// GraphAttnMultiHead forward for MI455X (gfx1250, wave32, WMMA)
// N=4096 nodes, IN_F=256, H=4 heads, D=64, H*D=256
// ---------------------------------------------------------------------------

#define NN   4096   // nodes
#define INF_ 256    // input features
#define HH   4      // heads
#define DD   64     // head dim
#define HD   256    // H*D
#define SLOPE 0.2f

typedef __attribute__((ext_vector_type(16))) __bf16 v16bf;
typedef __attribute__((ext_vector_type(8)))  float  v8f;

union BF16x16 { uint4 q[2]; v16bf v; };

__device__ __forceinline__ float wave_max(float v) {
#pragma unroll
    for (int o = 16; o > 0; o >>= 1) v = fmaxf(v, __shfl_xor(v, o, 32));
    return v;
}
__device__ __forceinline__ float wave_sum(float v) {
#pragma unroll
    for (int o = 16; o > 0; o >>= 1) v += __shfl_xor(v, o, 32);
    return v;
}

// --------------------------- K0: fp32 -> bf16 casts -------------------------
__global__ void k_convert(const float* __restrict__ inputs,
                          const float* __restrict__ weight,   // [INF_][HD]
                          const float* __restrict__ proj_w,   // [HD][INF_]
                          __bf16* __restrict__ in_bf,         // [NN][INF_]
                          __bf16* __restrict__ wT_bf,         // [HD][INF_] = weight^T
                          __bf16* __restrict__ pw_bf) {       // [HD][INF_] = proj_w
    int tid = blockIdx.x * blockDim.x + threadIdx.x;
    int stride = gridDim.x * blockDim.x;
    for (int i = tid; i < NN * INF_; i += stride)
        in_bf[i] = (__bf16)inputs[i];
    for (int i = tid; i < HD * INF_; i += stride) {
        int j = i / INF_, k = i - j * INF_;
        wT_bf[i] = (__bf16)weight[k * HD + j];   // transpose
        pw_bf[i] = (__bf16)proj_w[i];            // already [out][k]
    }
}

// ------------- K1: support = inputs @ weight (WMMA bf16, 16x16 tiles) -------
__global__ void __launch_bounds__(32)
k_gemm_support(const __bf16* __restrict__ A,    // [NN][INF_]
               const __bf16* __restrict__ BT,   // [HD][INF_]  (B transposed)
               float*  __restrict__ sup,        // [HH][NN][DD] fp32
               __bf16* __restrict__ supT) {     // [HH][DD][NN] bf16
    const int lane = threadIdx.x;
    const int i0 = blockIdx.x * 16;
    const int j0 = blockIdx.y * 16;
    const int r  = lane & 15;
    const int hi = lane >> 4;

    // A: lanes0-15 M=lane K0-7/16-23 ; lanes16-31 M=lane-16 K8-15/24-31
    const __bf16* arow = A  + (size_t)(i0 + r) * INF_ + hi * 8;
    // B: lanes0-15 K0-7/16-23 ; lanes16-31 K8-15/24-31 for column N = lane&15
    const __bf16* brow = BT + (size_t)(j0 + r) * INF_ + hi * 8;

    v8f acc = {};
#pragma unroll
    for (int k0 = 0; k0 < INF_; k0 += 32) {
        BF16x16 a, b;
        a.q[0] = *(const uint4*)(arow + k0);
        a.q[1] = *(const uint4*)(arow + k0 + 16);
        b.q[0] = *(const uint4*)(brow + k0);
        b.q[1] = *(const uint4*)(brow + k0 + 16);
        acc = __builtin_amdgcn_wmma_f32_16x16x32_bf16(false, a.v, false, b.v,
                                                      (short)0, acc, false, false);
    }
#pragma unroll
    for (int v = 0; v < 8; ++v) {
        int row = i0 + v + hi * 8;
        int col = j0 + r;
        int h = col >> 6, d = col & 63;
        sup[(((size_t)h * NN) + row) * DD + d]  = acc[v];
        supT[(((size_t)h * DD) + d) * NN + row] = (__bf16)acc[v];
    }
}

// --------------- K2: f1[h,n]=<sup,u>, f2[h,n]=<sup,v> (wave dots) -----------
__global__ void __launch_bounds__(32)
k_f1f2(const float* __restrict__ sup, const float* __restrict__ wu,
       const float* __restrict__ wv, float* __restrict__ f1, float* __restrict__ f2) {
    const int bid = blockIdx.x;            // h*NN + n
    const int h = bid >> 12;
    const int lane = threadIdx.x;
    const float* s = sup + (size_t)bid * DD;
    float s0 = s[lane], s1 = s[lane + 32];
    float r1 = wave_sum(s0 * wu[h * DD + lane] + s1 * wu[h * DD + lane + 32]);
    float r2 = wave_sum(s0 * wv[h * DD + lane] + s1 * wv[h * DD + lane + 32]);
    if (lane == 0) { f1[bid] = r1; f2[bid] = r2; }
}

// --------- K3: per-row sparse-softmax stats, ALL 4 heads share one adj pass -
// block = 128 threads = 4 waves; wave h handles head h for row i = blockIdx.x.
// The 4 waves read identical adj cachelines -> WGP$/L2 hits, ~1x HBM traffic.
__global__ void __launch_bounds__(128)
k_stats(const int* __restrict__ adj, const float* __restrict__ f1,
        const float* __restrict__ f2, float* __restrict__ rowmax,
        float* __restrict__ denom) {
    const int i    = blockIdx.x;
    const int h    = threadIdx.x >> 5;
    const int lane = threadIdx.x & 31;
    const int idx  = h * NN + i;
    const float f2i = f2[idx];
    const int*   arow = adj + (size_t)i * NN;
    const float* f1h  = f1 + h * NN;

    float mx = -__builtin_inff();
    for (int j = lane; j < NN; j += 32) {
        __builtin_prefetch(arow + j + 256, 0, 1);   // global_prefetch_b8
        float s = f1h[j] + f2i;
        s = s > 0.f ? s : SLOPE * s;
        float m = s * (float)arow[j];
        if (m != 0.f) mx = fmaxf(mx, m);
    }
    mx = wave_max(mx);
    if (mx == -__builtin_inff()) mx = 0.f;   // empty row

    float sum = 0.f;
    for (int j = lane; j < NN; j += 32) {
        float s = f1h[j] + f2i;
        s = s > 0.f ? s : SLOPE * s;
        float m = s * (float)arow[j];
        if (m != 0.f) sum += __expf(m - mx);
    }
    sum = wave_sum(sum);
    if (lane == 0) { rowmax[idx] = mx; denom[idx] = sum; }
}

// ---- K4: out = inputs @ proj_w.T + proj_b + bias (WMMA, writes d_out) ------
__global__ void __launch_bounds__(32)
k_proj(const __bf16* __restrict__ A, const __bf16* __restrict__ PW,
       const float* __restrict__ bias, const float* __restrict__ proj_b,
       float* __restrict__ out) {
    const int lane = threadIdx.x;
    const int i0 = blockIdx.x * 16;
    const int j0 = blockIdx.y * 16;
    const int r  = lane & 15;
    const int hi = lane >> 4;
    const __bf16* arow = A  + (size_t)(i0 + r) * INF_ + hi * 8;
    const __bf16* brow = PW + (size_t)(j0 + r) * INF_ + hi * 8;
    v8f acc = {};
#pragma unroll
    for (int k0 = 0; k0 < INF_; k0 += 32) {
        BF16x16 a, b;
        a.q[0] = *(const uint4*)(arow + k0);
        a.q[1] = *(const uint4*)(arow + k0 + 16);
        b.q[0] = *(const uint4*)(brow + k0);
        b.q[1] = *(const uint4*)(brow + k0 + 16);
        acc = __builtin_amdgcn_wmma_f32_16x16x32_bf16(false, a.v, false, b.v,
                                                      (short)0, acc, false, false);
    }
#pragma unroll
    for (int v = 0; v < 8; ++v) {
        int row = i0 + v + hi * 8;
        int col = j0 + r;
        out[(size_t)row * HD + col] = acc[v] + bias[col] + proj_b[col];
    }
}

// ------ K5: attention aggregation out += attn @ support (scores on-the-fly) -
// block = 128 threads = 4 waves; wave h handles head h for rows [i0, i0+16).
// All 4 waves walk the SAME 16x32 adj tile per j-step -> one HBM pass total.
// Each wave has a private LDS score tile + private WMMA accumulators.
__global__ void __launch_bounds__(128)
k_attn(const int* __restrict__ adj, const float* __restrict__ f1,
       const float* __restrict__ f2, const float* __restrict__ rowmax,
       const float* __restrict__ denom, const __bf16* __restrict__ supT,
       float* __restrict__ out) {
    __shared__ __attribute__((aligned(16))) __bf16 sA[HH][16 * 32]; // per-wave tile
    __shared__ float sF2[HH][16], sMX[HH][16], sRD[HH][16];
    const int h    = threadIdx.x >> 5;
    const int lane = threadIdx.x & 31;
    const int i0   = blockIdx.x * 16;
    const int r    = lane & 15;
    const int hi   = lane >> 4;

    if (lane < 16) {
        int idx = h * NN + i0 + lane;
        sF2[h][lane] = f2[idx];
        sMX[h][lane] = rowmax[idx];
        float dn = denom[idx];
        sRD[h][lane] = dn > 0.f ? 1.f / dn : 0.f;
    }
    __syncthreads();

    v8f acc[4] = {};
    const __bf16* supTh = supT + (size_t)h * DD * NN;
    __bf16* myA = sA[h];

    for (int j0 = 0; j0 < NN; j0 += 32) {
        // prefetch next adj j-block (lanes cover the 16 rows)
        __builtin_prefetch(&adj[(size_t)(i0 + r) * NN + j0 + 32 + hi * 16], 0, 1);
        // regenerate 16x32 normalized attention tile, lane = local column j
        float f1j = f1[h * NN + j0 + lane];
#pragma unroll 4
        for (int i = 0; i < 16; ++i) {
            int a = adj[(size_t)(i0 + i) * NN + j0 + lane];
            float s = f1j + sF2[h][i];
            s = s > 0.f ? s : SLOPE * s;
            float m = s * (float)a;
            float e = (m != 0.f) ? __expf(m - sMX[h][i]) : 0.f;
            myA[i * 32 + lane] = (__bf16)(e * sRD[h][i]);
        }
        __syncthreads();

        BF16x16 a;
        const __bf16* arow = myA + r * 32 + hi * 8;
        a.q[0] = *(const uint4*)(arow);
        a.q[1] = *(const uint4*)(arow + 16);
#pragma unroll
        for (int t = 0; t < 4; ++t) {        // 4 tiles cover D=64
            BF16x16 b;
            const __bf16* brow = supTh + (size_t)(t * 16 + r) * NN + j0 + hi * 8;
            b.q[0] = *(const uint4*)(brow);
            b.q[1] = *(const uint4*)(brow + 16);
            acc[t] = __builtin_amdgcn_wmma_f32_16x16x32_bf16(false, a.v, false, b.v,
                                                             (short)0, acc[t], false, false);
        }
        __syncthreads();
    }
#pragma unroll
    for (int t = 0; t < 4; ++t)
#pragma unroll
        for (int v = 0; v < 8; ++v) {
            int row = i0 + v + hi * 8;
            int col = h * DD + t * 16 + r;
            out[(size_t)row * HD + col] += acc[t][v];
        }
}

// ---------------------------------------------------------------------------
extern "C" void kernel_launch(void* const* d_in, const int* in_sizes, int n_in,
                              void* d_out, int out_size, void* d_ws, size_t ws_size,
                              hipStream_t stream) {
    (void)in_sizes; (void)n_in; (void)out_size; (void)ws_size;
    const float* inputs = (const float*)d_in[0];
    const int*   adj    = (const int*)d_in[1];
    const float* weight = (const float*)d_in[2];
    const float* wu     = (const float*)d_in[3];
    const float* wv     = (const float*)d_in[4];
    const float* bias   = (const float*)d_in[5];
    const float* proj_w = (const float*)d_in[6];
    const float* proj_b = (const float*)d_in[7];
    float* out = (float*)d_out;

    char* ws = (char*)d_ws;
    auto take = [&](size_t bytes) -> char* {
        char* p = ws; ws += (bytes + 255) & ~(size_t)255; return p;
    };
    __bf16* in_bf  = (__bf16*)take((size_t)NN * INF_ * 2);   // 2 MB
    __bf16* wT_bf  = (__bf16*)take((size_t)HD * INF_ * 2);   // 128 KB
    __bf16* pw_bf  = (__bf16*)take((size_t)HD * INF_ * 2);   // 128 KB
    float*  sup    = (float*) take((size_t)HH * NN * DD * 4);// 4 MB
    __bf16* supT   = (__bf16*)take((size_t)HH * DD * NN * 2);// 2 MB
    float*  f1     = (float*) take((size_t)HH * NN * 4);
    float*  f2     = (float*) take((size_t)HH * NN * 4);
    float*  rowmax = (float*) take((size_t)HH * NN * 4);
    float*  denom  = (float*) take((size_t)HH * NN * 4);

    k_convert<<<2048, 256, 0, stream>>>(inputs, weight, proj_w, in_bf, wT_bf, pw_bf);
    k_gemm_support<<<dim3(NN / 16, HD / 16), 32, 0, stream>>>(in_bf, wT_bf, sup, supT);
    k_f1f2<<<HH * NN, 32, 0, stream>>>(sup, wu, wv, f1, f2);
    k_stats<<<NN, 128, 0, stream>>>(adj, f1, f2, rowmax, denom);
    k_proj<<<dim3(NN / 16, HD / 16), 32, 0, stream>>>(in_bf, pw_bf, bias, proj_b, out);
    k_attn<<<dim3(NN / 16), 128, 0, stream>>>(adj, f1, f2, rowmax, denom, supT, out);
}
